// MegaCRN_81441169866972
// MI455X (gfx1250) — matrix operand order repaired and checked by reading, not compiled, not run on hardware
//
#include <hip/hip_runtime.h>
#include <math.h>
#include <stdint.h>

typedef __bf16 bhalf;
typedef __attribute__((ext_vector_type(16))) __bf16 v16bf;
typedef __attribute__((ext_vector_type(8)))  float  v8f;
typedef int v4i __attribute__((vector_size(16)));

#if defined(__has_builtin)
#if __has_builtin(__builtin_amdgcn_global_load_async_to_lds_b128) && \
    __has_builtin(__builtin_amdgcn_s_wait_asynccnt)
#define USE_ASYNC_LDS 1
#endif
#endif

#define NNODE 1024
#define BATCH 64
#define TSTEP 12
#define HSTEP 12
#define RU    64
#define MEMN  20
#define MEMD  64
#define DECD  128
#define NB    (NNODE*BATCH)
#define EC    (1+RU)     // 65  encoder concat width
#define DC    (2+DECD)   // 130 decoder concat width

#define FLAG_ACC  1
#define FLAG_CHEB 2

#define LDP 40   // padded LDS row stride (80B: bank-conflict-free, 16B aligned)

// ---------------------------------------------------------------------------
// Generic bf16 WMMA GEMM: C[M,N] = A[M,K] @ B[K,N] (+C if FLAG_ACC)
// Block tile 128x64, 8 waves; each wave owns a 32x32 quadrant (2x2 wmma tiles).
// Double-buffered LDS staging; A staged with async b128->LDS (gfx1250) when
// aligned; B staged transposed ([col][k]) so fragment reads are ds_load_b128.
// Epilogue: write fp32 (Cf) or bf16 (Cb); FLAG_CHEB applies 2*x - I.
// ---------------------------------------------------------------------------
__global__ __launch_bounds__(256)
void gemm_bf16_kernel(const bhalf* __restrict__ A, int lda,
                      const bhalf* __restrict__ Bm, int ldb,
                      float* __restrict__ Cf, bhalf* __restrict__ Cb, int ldc,
                      int M, int Ncol, int K, int flags)
{
    __shared__ bhalf As[2][128][LDP];
    __shared__ bhalf Bs[2][64][LDP];   // transposed: [col][k]

    const int tid   = threadIdx.x;
    const int lane  = tid & 31;
    const int wave  = tid >> 5;
    const int m0    = blockIdx.y * 128;
    const int n0    = blockIdx.x * 64;
    const int wm    = (wave >> 1) * 32;   // 0,32,64,96
    const int wn    = (wave & 1) * 32;    // 0,32
    const int lhalf = lane >> 4;          // 0 or 1
    const int l16   = lane & 15;

    const bool mfull = (m0 + 128 <= M);
    const bool nfull = (n0 + 64 <= Ncol);
    const bool avec  = mfull && ((lda & 7) == 0) && ((((uintptr_t)A)  & 15) == 0);
    const bool bvec  = nfull && ((ldb & 7) == 0) && ((((uintptr_t)Bm) & 15) == 0);

    auto stageA = [&](int k0, int bi) {
        const bool kfull = (k0 + 32 <= K);
        if (avec && kfull) {
            #pragma unroll
            for (int it = 0; it < 2; it++) {
                int e = tid + it*256;                 // 0..511 chunks of 8 bf16
                int r = e >> 2, kk = (e & 3) * 8;
                const bhalf* src = A + (size_t)(m0 + r)*lda + (k0 + kk);
                bhalf*       dst = &As[bi][r][kk];
#ifdef USE_ASYNC_LDS
                __builtin_amdgcn_global_load_async_to_lds_b128((v4i*)src, (v4i*)dst, 0, 0);
#else
                *(uint4*)dst = *(const uint4*)src;
#endif
            }
        } else if (mfull && kfull) {
            #pragma unroll
            for (int it = 0; it < 16; it++) {
                int e = tid + it*256;                 // 0..4095, unguarded
                int r = e >> 5, kk = e & 31;
                As[bi][r][kk] = A[(size_t)(m0 + r)*lda + (k0 + kk)];
            }
        } else {
            for (int e = tid; e < 128*32; e += 256) {
                int r = e >> 5, kk = e & 31;
                int gm = m0 + r, gk = k0 + kk;
                As[bi][r][kk] = (gm < M && gk < K) ? A[(size_t)gm*lda + gk] : (bhalf)0.f;
            }
        }
    };

    auto stageB = [&](int k0, int bi) {
        const bool kfull = (k0 + 32 <= K);
        if (bvec && kfull) {
            int r = tid >> 3, c = (tid & 7) * 8;      // row k, 8 cols per thread
            uint4 v = *(const uint4*)(Bm + (size_t)(k0 + r)*ldb + (n0 + c));
            const bhalf* pv = (const bhalf*)&v;
            #pragma unroll
            for (int i = 0; i < 8; i++) Bs[bi][c + i][r] = pv[i];
        } else {
            for (int e = tid; e < 32*64; e += 256) {
                int r = e >> 6, c = e & 63;
                int gk = k0 + r, gn = n0 + c;
                Bs[bi][c][r] = (gk < K && gn < Ncol) ? Bm[(size_t)gk*ldb + gn] : (bhalf)0.f;
            }
        }
    };

    v8f acc[2][2];
    #pragma unroll
    for (int i = 0; i < 2; i++)
      #pragma unroll
      for (int j = 0; j < 2; j++)
        #pragma unroll
        for (int e = 0; e < 8; e++) acc[i][j][e] = 0.f;

    if (flags & FLAG_ACC) {
      #pragma unroll
      for (int i = 0; i < 2; i++)
        #pragma unroll
        for (int j = 0; j < 2; j++)
          #pragma unroll
          for (int e = 0; e < 8; e++) {
            int gm = m0 + wm + 16*i + (e + 8*lhalf);
            int gn = n0 + wn + 16*j + l16;
            acc[i][j][e] = (gm < M && gn < Ncol) ? Cf[(size_t)gm*ldc + gn] : 0.f;
          }
    }

    const int nk = (K + 31) / 32;
    stageA(0, 0);
    stageB(0, 0);
#ifdef USE_ASYNC_LDS
    __builtin_amdgcn_s_wait_asynccnt(0);
#endif
    __syncthreads();

    for (int kc = 0; kc < nk; kc++) {
        const int bi = kc & 1;
        if (kc + 1 < nk) {                 // prefetch next chunk into other buffer
            stageA((kc + 1) * 32, 1 - bi);
            stageB((kc + 1) * 32, 1 - bi);
        }

        // A fragment: lane l16 = row; VGPR pair p covers K = (p<4?0:16)+8*lhalf+2*(p&3)
        v16bf af[2], bfr[2];
        #pragma unroll
        for (int i = 0; i < 2; i++) {
            int row = wm + 16*i + l16;
            #pragma unroll
            for (int p = 0; p < 8; p++) {
                int kk = ((p < 4) ? 0 : 16) + 8*lhalf + 2*(p & 3);
                af[i][2*p]   = As[bi][row][kk];
                af[i][2*p+1] = As[bi][row][kk+1];
            }
        }
        // B fragment (transposed LDS): lane l16 = col; same K layout as A fragment
        #pragma unroll
        for (int j = 0; j < 2; j++) {
            int col = wn + 16*j + l16;
            #pragma unroll
            for (int p = 0; p < 8; p++) {
                int kk = ((p < 4) ? 0 : 16) + 8*lhalf + 2*(p & 3);
                bfr[j][2*p]   = Bs[bi][col][kk];
                bfr[j][2*p+1] = Bs[bi][col][kk+1];
            }
        }
        #pragma unroll
        for (int i = 0; i < 2; i++)
            #pragma unroll
            for (int j = 0; j < 2; j++)
                acc[i][j] = __builtin_amdgcn_wmma_f32_16x16x32_bf16(
                    false, af[i], false, bfr[j], (short)0, acc[i][j], false, false);

#ifdef USE_ASYNC_LDS
        __builtin_amdgcn_s_wait_asynccnt(0);
#endif
        __syncthreads();
    }

    #pragma unroll
    for (int i = 0; i < 2; i++)
      #pragma unroll
      for (int j = 0; j < 2; j++)
        #pragma unroll
        for (int e = 0; e < 8; e++) {
          int gm = m0 + wm + 16*i + (e + 8*lhalf);
          int gn = n0 + wn + 16*j + l16;
          if (gm < M && gn < Ncol) {
            float v = acc[i][j][e];
            if (flags & FLAG_CHEB) v = 2.f*v - ((gm == gn) ? 1.f : 0.f);
            if (Cb) Cb[(size_t)gm*ldc + gn] = (bhalf)v;
            else    Cf[(size_t)gm*ldc + gn] = v;
          }
        }
}

// ---------------------------------------------------------------------------
// Small / elementwise kernels
// ---------------------------------------------------------------------------
__global__ void negemm_kernel(float* __restrict__ ne, const float* __restrict__ We,
                              const float* __restrict__ Mem)
{
    int idx = blockIdx.x * blockDim.x + threadIdx.x;
    if (idx >= NNODE * MEMD) return;
    int n = idx / MEMD, d = idx % MEMD;
    float s = 0.f;
    #pragma unroll
    for (int k = 0; k < MEMN; k++) s += We[n*MEMN + k] * Mem[k*MEMD + d];
    ne[idx] = s;
}

__global__ void scores_kernel(float* __restrict__ S, const float* __restrict__ nea,
                              const float* __restrict__ neb)
{
    int idx = blockIdx.x * blockDim.x + threadIdx.x;
    if (idx >= NNODE * NNODE) return;
    int n = idx / NNODE, m = idx % NNODE;
    float s = 0.f;
    #pragma unroll 8
    for (int d = 0; d < MEMD; d++) s += nea[n*MEMD + d] * neb[m*MEMD + d];
    S[idx] = fmaxf(s, 0.f);
}

__global__ void softmax_row_kernel(bhalf* __restrict__ g, const float* __restrict__ S)
{
    __shared__ float red[256];
    int n = blockIdx.x, tid = threadIdx.x;
    const float* row = S + (size_t)n * NNODE;
    float mx = -1e30f;
    for (int m = tid; m < NNODE; m += 256) mx = fmaxf(mx, row[m]);
    red[tid] = mx; __syncthreads();
    for (int s = 128; s > 0; s >>= 1) { if (tid < s) red[tid] = fmaxf(red[tid], red[tid+s]); __syncthreads(); }
    mx = red[0]; __syncthreads();
    float sum = 0.f;
    for (int m = tid; m < NNODE; m += 256) sum += expf(row[m] - mx);
    red[tid] = sum; __syncthreads();
    for (int s = 128; s > 0; s >>= 1) { if (tid < s) red[tid] += red[tid+s]; __syncthreads(); }
    float inv = 1.f / red[0];
    for (int m = tid; m < NNODE; m += 256)
        g[(size_t)n*NNODE + m] = (bhalf)(expf(row[m] - mx) * inv);
}

__global__ void cvt_bf_kernel(bhalf* __restrict__ dst, const float* __restrict__ src, int n)
{
    int i = blockIdx.x * blockDim.x + threadIdx.x;
    if (i < n) dst[i] = (bhalf)src[i];
}

__global__ void zero_kernel(float* __restrict__ x, int n)
{
    int i = blockIdx.x * blockDim.x + threadIdx.x;
    if (i < n) x[i] = 0.f;
}

__global__ void fill_bias_kernel(float* __restrict__ out, const float* __restrict__ bias,
                                 int rows, int O)
{
    int i = blockIdx.x * blockDim.x + threadIdx.x;
    if (i < rows * O) out[i] = bias[i % O];
}

__global__ void act_kernel(float* __restrict__ x, int n, int mode)
{
    int i = blockIdx.x * blockDim.x + threadIdx.x;
    if (i >= n) return;
    float v = x[i];
    x[i] = mode ? tanhf(v) : 1.f / (1.f + expf(-v));
}

__global__ void gru_kernel(float* __restrict__ h, const float* __restrict__ zr,
                           const float* __restrict__ hc, int O)
{
    int i = blockIdx.x * blockDim.x + threadIdx.x;
    if (i >= NB * O) return;
    int nb = i / O, u = i % O;
    float r = zr[(size_t)nb*2*O + O + u];
    h[i] = r * h[i] + (1.f - r) * hc[i];
}

__global__ void build_inp_enc(bhalf* __restrict__ X, const float* __restrict__ hist,
                              const float* __restrict__ hE, int t)
{
    int idx = blockIdx.x * blockDim.x + threadIdx.x;
    if (idx >= NB * EC) return;
    int c = idx % EC, nb = idx / EC;
    int n = nb / BATCH, b = nb % BATCH;
    float v = (c == 0) ? hist[((size_t)b*TSTEP + t)*NNODE + n]
                       : hE[(size_t)nb*RU + (c-1)];
    X[idx] = (bhalf)v;
}

__global__ void build_cand_enc(bhalf* __restrict__ X, const float* __restrict__ hist,
                               const float* __restrict__ hE, const float* __restrict__ zr, int t)
{
    int idx = blockIdx.x * blockDim.x + threadIdx.x;
    if (idx >= NB * EC) return;
    int c = idx % EC, nb = idx / EC;
    int n = nb / BATCH, b = nb % BATCH;
    float v;
    if (c == 0) v = hist[((size_t)b*TSTEP + t)*NNODE + n];
    else        v = zr[(size_t)nb*2*RU + (c-1)] * hE[(size_t)nb*RU + (c-1)];
    X[idx] = (bhalf)v;
}

__global__ void build_inp_dec(bhalf* __restrict__ X, const float* __restrict__ go,
                              const float* __restrict__ fut, const float* __restrict__ hD, int t)
{
    int idx = blockIdx.x * blockDim.x + threadIdx.x;
    if (idx >= NB * DC) return;
    int c = idx % DC, nb = idx / DC;
    int n = nb / BATCH, b = nb % BATCH;
    float v;
    if (c == 0)      v = go[nb];
    else if (c == 1) v = fut[(((size_t)b*HSTEP + t)*NNODE + n)*2 + 1];
    else             v = hD[(size_t)nb*DECD + (c-2)];
    X[idx] = (bhalf)v;
}

__global__ void build_cand_dec(bhalf* __restrict__ X, const float* __restrict__ go,
                               const float* __restrict__ fut, const float* __restrict__ zr,
                               const float* __restrict__ hD, int t)
{
    int idx = blockIdx.x * blockDim.x + threadIdx.x;
    if (idx >= NB * DC) return;
    int c = idx % DC, nb = idx / DC;
    int n = nb / BATCH, b = nb % BATCH;
    float v;
    if (c == 0)      v = go[nb];
    else if (c == 1) v = fut[(((size_t)b*HSTEP + t)*NNODE + n)*2 + 1];
    else             v = zr[(size_t)nb*2*DECD + (c-2)] * hD[(size_t)nb*DECD + (c-2)];
    X[idx] = (bhalf)v;
}

__global__ void attn_kernel(const float* __restrict__ hE, const float* __restrict__ Wq,
                            const float* __restrict__ Mem, float* __restrict__ hD)
{
    int nb = blockIdx.x * blockDim.x + threadIdx.x;
    if (nb >= NB) return;
    float hh[RU];
    #pragma unroll 8
    for (int u = 0; u < RU; u++) hh[u] = hE[(size_t)nb*RU + u];
    float q[MEMD];
    for (int d = 0; d < MEMD; d++) {
        float s = 0.f;
        #pragma unroll 8
        for (int u = 0; u < RU; u++) s += hh[u] * Wq[u*MEMD + d];
        q[d] = s;
    }
    float att[MEMN]; float mx = -1e30f;
    for (int p = 0; p < MEMN; p++) {
        float s = 0.f;
        #pragma unroll 8
        for (int d = 0; d < MEMD; d++) s += q[d] * Mem[p*MEMD + d];
        att[p] = s; mx = fmaxf(mx, s);
    }
    float sum = 0.f;
    for (int p = 0; p < MEMN; p++) { att[p] = expf(att[p] - mx); sum += att[p]; }
    float inv = 1.f / sum;
    for (int u = 0; u < RU; u++) hD[(size_t)nb*DECD + u] = hh[u];
    for (int d = 0; d < MEMD; d++) {
        float v = 0.f;
        for (int p = 0; p < MEMN; p++) v += att[p] * Mem[p*MEMD + d];
        hD[(size_t)nb*DECD + RU + d] = v * inv;
    }
}

__global__ void proj_kernel(const float* __restrict__ hD, const float* __restrict__ pw,
                            const float* __restrict__ pb, float* __restrict__ go,
                            float* __restrict__ out, int t)
{
    int nb = blockIdx.x * blockDim.x + threadIdx.x;
    if (nb >= NB) return;
    int n = nb / BATCH, b = nb % BATCH;
    float s = pb[0];
    #pragma unroll 8
    for (int d = 0; d < DECD; d++) s += hD[(size_t)nb*DECD + d] * pw[d];
    go[nb] = s;
    out[((size_t)b*HSTEP + t)*NNODE + n] = s;
}

// ---------------------------------------------------------------------------
// Host side
// ---------------------------------------------------------------------------
static void launch_gemm(const bhalf* A, int lda, const bhalf* Bm, int ldb,
                        float* Cf, bhalf* Cb, int ldc, int M, int Ncol, int K,
                        int flags, hipStream_t s)
{
    dim3 grid((Ncol + 63) / 64, (M + 127) / 128), block(256);
    gemm_bf16_kernel<<<grid, block, 0, s>>>(A, lda, Bm, ldb, Cf, Cb, ldc, M, Ncol, K, flags);
}

static inline dim3 ew(int n) { return dim3((n + 255) / 256); }

// agcn: outf(NB,O) = bias + sum_j (s_j @ X) @ W_j   (supports: I, g1, S2, I, g2, S5)
static void agcn(const bhalf* X, bhalf* Y,
                 const bhalf* g1b, const bhalf* s2b, const bhalf* g2b, const bhalf* s5b,
                 const bhalf* Wbf, const float* bias, float* outf,
                 int C, int O, hipStream_t s)
{
    int BC = BATCH * C;
    fill_bias_kernel<<<ew(NB*O), 256, 0, s>>>(outf, bias, NB, O);
    const bhalf* sup[6] = {nullptr, g1b, s2b, nullptr, g2b, s5b};
    for (int j = 0; j < 6; j++) {
        const bhalf* Aj;
        if (sup[j]) {
            // Y(N, B*C) = s_j(N,N) @ X(N, B*C); row-major (N,B*C) == (N*B, C)
            launch_gemm(sup[j], NNODE, X, BC, nullptr, Y, BC, NNODE, BC, NNODE, 0, s);
            Aj = Y;
        } else Aj = X;
        launch_gemm(Aj, C, Wbf + (size_t)j*C*O, O, outf, nullptr, O, NB, O, C, FLAG_ACC, s);
    }
}

extern "C" void kernel_launch(void* const* d_in, const int* in_sizes, int n_in,
                              void* d_out, int out_size, void* d_ws, size_t ws_size,
                              hipStream_t stream)
{
    (void)in_sizes; (void)n_in; (void)out_size; (void)ws_size;
    const float* history = (const float*)d_in[0];
    const float* future  = (const float*)d_in[1];
    const float* Memory  = (const float*)d_in[2];
    const float* Wq      = (const float*)d_in[3];
    const float* We1     = (const float*)d_in[4];
    const float* We2     = (const float*)d_in[5];
    const float* egw     = (const float*)d_in[6];
    const float* egb     = (const float*)d_in[7];
    const float* euw     = (const float*)d_in[8];
    const float* eub     = (const float*)d_in[9];
    const float* dgw     = (const float*)d_in[10];
    const float* dgb     = (const float*)d_in[11];
    const float* duw     = (const float*)d_in[12];
    const float* dub     = (const float*)d_in[13];
    const float* pw      = (const float*)d_in[14];
    const float* pb      = (const float*)d_in[15];
    float* out = (float*)d_out;

    char* base = (char*)d_ws;
    size_t off = 0;
    auto alloc = [&](size_t bytes) -> void* {
        void* p = base + off;
        off += (bytes + 255) & ~(size_t)255;
        return p;
    };

    float* ne1 = (float*)alloc((size_t)NNODE*MEMD*4);
    float* ne2 = (float*)alloc((size_t)NNODE*MEMD*4);
    float* Sc  = (float*)alloc((size_t)NNODE*NNODE*4);
    bhalf* g1b = (bhalf*)alloc((size_t)NNODE*NNODE*2);
    bhalf* g2b = (bhalf*)alloc((size_t)NNODE*NNODE*2);
    bhalf* s2b = (bhalf*)alloc((size_t)NNODE*NNODE*2);
    bhalf* s5b = (bhalf*)alloc((size_t)NNODE*NNODE*2);
    float* hE  = (float*)alloc((size_t)NB*RU*4);
    float* hD  = (float*)alloc((size_t)NB*DECD*4);
    bhalf* X   = (bhalf*)alloc((size_t)NB*DC*2);
    bhalf* Y   = (bhalf*)alloc((size_t)NB*DC*2);
    float* zr  = (float*)alloc((size_t)NB*2*DECD*4);
    float* hc  = (float*)alloc((size_t)NB*DECD*4);
    float* go  = (float*)alloc((size_t)NB*4);
    bhalf* egwb = (bhalf*)alloc((size_t)6*EC*2*RU*2);
    bhalf* euwb = (bhalf*)alloc((size_t)6*EC*RU*2);
    bhalf* dgwb = (bhalf*)alloc((size_t)6*DC*2*DECD*2);
    bhalf* duwb = (bhalf*)alloc((size_t)6*DC*DECD*2);

    // Graph construction
    negemm_kernel<<<ew(NNODE*MEMD), 256, 0, stream>>>(ne1, We1, Memory);
    negemm_kernel<<<ew(NNODE*MEMD), 256, 0, stream>>>(ne2, We2, Memory);
    scores_kernel<<<ew(NNODE*NNODE), 256, 0, stream>>>(Sc, ne1, ne2);
    softmax_row_kernel<<<NNODE, 256, 0, stream>>>(g1b, Sc);
    scores_kernel<<<ew(NNODE*NNODE), 256, 0, stream>>>(Sc, ne2, ne1);
    softmax_row_kernel<<<NNODE, 256, 0, stream>>>(g2b, Sc);
    // Chebyshev order-2 supports: S = 2*g@g - I  (WMMA, bf16 out)
    launch_gemm(g1b, NNODE, g1b, NNODE, nullptr, s2b, NNODE, NNODE, NNODE, NNODE, FLAG_CHEB, stream);
    launch_gemm(g2b, NNODE, g2b, NNODE, nullptr, s5b, NNODE, NNODE, NNODE, NNODE, FLAG_CHEB, stream);

    // Weights to bf16
    cvt_bf_kernel<<<ew(6*EC*2*RU), 256, 0, stream>>>(egwb, egw, 6*EC*2*RU);
    cvt_bf_kernel<<<ew(6*EC*RU),   256, 0, stream>>>(euwb, euw, 6*EC*RU);
    cvt_bf_kernel<<<ew(6*DC*2*DECD), 256, 0, stream>>>(dgwb, dgw, 6*DC*2*DECD);
    cvt_bf_kernel<<<ew(6*DC*DECD),   256, 0, stream>>>(duwb, duw, 6*DC*DECD);

    zero_kernel<<<ew(NB*RU), 256, 0, stream>>>(hE, NB*RU);

    // Encoder
    for (int t = 0; t < TSTEP; t++) {
        build_inp_enc<<<ew(NB*EC), 256, 0, stream>>>(X, history, hE, t);
        agcn(X, Y, g1b, s2b, g2b, s5b, egwb, egb, zr, EC, 2*RU, stream);
        act_kernel<<<ew(NB*2*RU), 256, 0, stream>>>(zr, NB*2*RU, 0);
        build_cand_enc<<<ew(NB*EC), 256, 0, stream>>>(X, history, hE, zr, t);
        agcn(X, Y, g1b, s2b, g2b, s5b, euwb, eub, hc, EC, RU, stream);
        act_kernel<<<ew(NB*RU), 256, 0, stream>>>(hc, NB*RU, 1);
        gru_kernel<<<ew(NB*RU), 256, 0, stream>>>(hE, zr, hc, RU);
    }

    // Memory attention -> decoder initial state h_de = [h, value]
    attn_kernel<<<ew(NB), 256, 0, stream>>>(hE, Wq, Memory, hD);
    zero_kernel<<<ew(NB), 256, 0, stream>>>(go, NB);

    // Decoder
    for (int t = 0; t < HSTEP; t++) {
        build_inp_dec<<<ew(NB*DC), 256, 0, stream>>>(X, go, future, hD, t);
        agcn(X, Y, g1b, s2b, g2b, s5b, dgwb, dgb, zr, DC, 2*DECD, stream);
        act_kernel<<<ew(NB*2*DECD), 256, 0, stream>>>(zr, NB*2*DECD, 0);
        build_cand_dec<<<ew(NB*DC), 256, 0, stream>>>(X, go, future, zr, hD, t);
        agcn(X, Y, g1b, s2b, g2b, s5b, duwb, dub, hc, DC, DECD, stream);
        act_kernel<<<ew(NB*DECD), 256, 0, stream>>>(hc, NB*DECD, 1);
        gru_kernel<<<ew(NB*DECD), 256, 0, stream>>>(hD, zr, hc, DECD);
        proj_kernel<<<ew(NB), 256, 0, stream>>>(hD, pw, pb, go, out, t);
    }
}
